// LinearAttention_5257039970858
// MI455X (gfx1250) — compile-verified, matrix operand order checked by reading, not run on hardware
//
#include <hip/hip_runtime.h>
#include <hip/hip_bf16.h>

typedef __attribute__((ext_vector_type(16))) _Float16 v16h;
typedef __attribute__((ext_vector_type(8)))  float    v8f;

#define BATCH   16
#define C_      256
#define HEADS   4
#define DH      32
#define HID     128
#define F3      384
#define NTOK    4096
#define NROWS   (BATCH * NTOK)          // 65536
#define QSCALE  0.17677669529663687f    // 32^-0.5
#define LN_EPS  1e-3f

// VGPR->K map for 16-bit A/B fragments of v_wmma_*_16x16x32 (ISA 7.12.2):
// lanes 0-15 hold K-half 0 (K=0..7 in V0-3, K=16..23 in V4-7),
// lanes 16-31 hold K-half 1 (K=8..15, K=24..31).
__device__ __forceinline__ int kmap16(int j, int khalf) {
  return (j < 8) ? (khalf * 8 + j) : (16 + khalf * 8 + (j - 8));
}

// Vectorized A-fragment load from a row-major f32 matrix row (stride `ld`),
// K window [k0, k0+32). Uses float4 loads (32B-aligned when k0%8==0).
__device__ __forceinline__ v16h load_a_frag(const float* __restrict__ rowptr,
                                            int k0, int khalf) {
  const float4* p0 = (const float4*)(rowptr + k0 + khalf * 8);
  const float4* p1 = (const float4*)(rowptr + k0 + 16 + khalf * 8);
  const float4 q0 = p0[0], q1 = p0[1], q2 = p1[0], q3 = p1[1];
  v16h a;
  a[0] = (_Float16)q0.x;  a[1] = (_Float16)q0.y;
  a[2] = (_Float16)q0.z;  a[3] = (_Float16)q0.w;
  a[4] = (_Float16)q1.x;  a[5] = (_Float16)q1.y;
  a[6] = (_Float16)q1.z;  a[7] = (_Float16)q1.w;
  a[8] = (_Float16)q2.x;  a[9] = (_Float16)q2.y;
  a[10] = (_Float16)q2.z; a[11] = (_Float16)q2.w;
  a[12] = (_Float16)q3.x; a[13] = (_Float16)q3.y;
  a[14] = (_Float16)q3.z; a[15] = (_Float16)q3.w;
  return a;
}

// ---------------- K1: qkv = X[65536,256] @ Wqkv[256,384] ----------------
// One wave computes a 16x64 output strip: A fragment loaded once per k-step
// (b128 loads), reused by 4 WMMAs.
__global__ void la_qkv_gemm(const float* __restrict__ x,
                            const float* __restrict__ w,
                            float* __restrict__ qkv) {
  const int lane  = threadIdx.x;
  const int mr    = lane & 15;
  const int khalf = lane >> 4;
  const int rowbase  = blockIdx.x * 16;
  const int colbase0 = blockIdx.y * 64;
  const float* arow = x + (size_t)(rowbase + mr) * C_;
  v8f acc0 = {}, acc1 = {}, acc2 = {}, acc3 = {};
  for (int k0 = 0; k0 < C_; k0 += 32) {
    const v16h a = load_a_frag(arow, k0, khalf);
    v16h b0, b1, b2, b3;
#pragma unroll
    for (int j = 0; j < 16; ++j) {
      const size_t woff = (size_t)(k0 + kmap16(j, khalf)) * F3 + colbase0 + mr;
      b0[j] = (_Float16)w[woff];
      b1[j] = (_Float16)w[woff + 16];
      b2[j] = (_Float16)w[woff + 32];
      b3[j] = (_Float16)w[woff + 48];
    }
    acc0 = __builtin_amdgcn_wmma_f32_16x16x32_f16(false, a, false, b0, (short)0, acc0, false, false);
    acc1 = __builtin_amdgcn_wmma_f32_16x16x32_f16(false, a, false, b1, (short)0, acc1, false, false);
    acc2 = __builtin_amdgcn_wmma_f32_16x16x32_f16(false, a, false, b2, (short)0, acc2, false, false);
    acc3 = __builtin_amdgcn_wmma_f32_16x16x32_f16(false, a, false, b3, (short)0, acc3, false, false);
  }
#pragma unroll
  for (int r = 0; r < 8; ++r) {
    const size_t row = (size_t)(rowbase + r + 8 * khalf) * F3 + colbase0 + mr;
    qkv[row]      = acc0[r];
    qkv[row + 16] = acc1[r];
    qkv[row + 32] = acc2[r];
    qkv[row + 48] = acc3[r];
  }
}

// ---------------- K2: k-softmax stats over n per (b,h,c) ----------------
__global__ void la_ksm_stats(const float* __restrict__ qkv,
                             float* __restrict__ kmax,
                             float* __restrict__ ksum) {
  const int bh = blockIdx.x;
  const int b  = bh / HEADS, h = bh % HEADS;
  const int c  = threadIdx.x & 31;
  const int g  = threadIdx.x >> 5;      // 0..7
  const float* base = qkv + (size_t)b * NTOK * F3 + 128 + h * DH + c;
  __shared__ float redm[8][32];
  __shared__ float reds[8][32];

  float m = -1e30f;
  for (int t = g; t < NTOK; t += 8)
    m = fmaxf(m, base[(size_t)t * F3]);
  redm[g][c] = m;
  __syncthreads();
#pragma unroll
  for (int i = 0; i < 8; ++i) m = fmaxf(m, redm[i][c]);

  float s = 0.f;
  for (int t = g; t < NTOK; t += 8)
    s += __expf(base[(size_t)t * F3] - m);
  reds[g][c] = s;
  __syncthreads();
  if (g == 0) {
    s = 0.f;
#pragma unroll
    for (int i = 0; i < 8; ++i) s += reds[i][c];
    const int idx = bh * DH + c;
    kmax[idx] = m;
    ksum[idx] = s;
  }
}

// ---------------- K3: ctx[b,h] = softmax(k)' ^T x V ----------------
// grid: (bh=64, dtile=2); wave computes a 16x32 strip: normalized-k A
// fragment (the expensive exp path) shared by both e-tiles.
__global__ void la_context_gemm(const float* __restrict__ qkv,
                                const float* __restrict__ kmax,
                                const float* __restrict__ ksum,
                                float* __restrict__ ctx) {
  const int bh = blockIdx.x;
  const int dt = blockIdx.y;
  const int b  = bh / HEADS, h = bh % HEADS;
  const int lane = threadIdx.x;
  const int mr = lane & 15, khalf = lane >> 4;
  const int d = dt * 16 + mr;           // A row (M)
  const int sidx = bh * DH + d;
  const float km   = kmax[sidx];
  const float kinv = 1.f / ksum[sidx];
  const float* rowb = qkv + (size_t)b * NTOK * F3;
  v8f acc0 = {}, acc1 = {};
  for (int n0 = 0; n0 < NTOK; n0 += 32) {
    v16h a, b0, b1;
#pragma unroll
    for (int j = 0; j < 16; ++j) {
      const int n = n0 + kmap16(j, khalf);
      const float kv = rowb[(size_t)n * F3 + 128 + h * DH + d];
      a[j]  = (_Float16)(__expf(kv - km) * kinv);                       // k'
      const size_t voff = (size_t)n * F3 + 256 + h * DH + mr;
      b0[j] = (_Float16)rowb[voff];                                     // v, e=mr
      b1[j] = (_Float16)rowb[voff + 16];                                // v, e=16+mr
    }
    acc0 = __builtin_amdgcn_wmma_f32_16x16x32_f16(false, a, false, b0, (short)0, acc0, false, false);
    acc1 = __builtin_amdgcn_wmma_f32_16x16x32_f16(false, a, false, b1, (short)0, acc1, false, false);
  }
#pragma unroll
  for (int r = 0; r < 8; ++r) {
    const int dd = dt * 16 + r + 8 * khalf;
    const size_t base = (size_t)bh * (DH * DH) + dd * DH + mr;
    ctx[base]      = acc0[r];
    ctx[base + 16] = acc1[r];
  }
}

// ---------------- K4: O[n, h*32+e] = (softmax_d(q)*scale) @ ctx ----------------
// grid: (token tiles=256, bh=64); softmax A fragment computed once, shared
// by both e-tiles; K=32 -> exactly one WMMA per tile.
__global__ void la_attn_out(const float* __restrict__ qkv,
                            const float* __restrict__ ctx,
                            float* __restrict__ O) {
  const int tt = blockIdx.x;
  const int bh = blockIdx.y;
  const int b  = bh / HEADS, h = bh % HEADS;
  const int lane = threadIdx.x;
  const int mr = lane & 15, khalf = lane >> 4;
  const int tok = tt * 16 + mr;
  const float* qrow = qkv + (size_t)(b * NTOK + tok) * F3 + h * DH;
  // f32 softmax over the 32 feature dims of this token (vectorized loads)
  float qv[DH];
#pragma unroll
  for (int d4 = 0; d4 < DH; d4 += 4) {
    const float4 q4 = *(const float4*)(qrow + d4);
    qv[d4] = q4.x; qv[d4 + 1] = q4.y; qv[d4 + 2] = q4.z; qv[d4 + 3] = q4.w;
  }
  float m = -1e30f;
#pragma unroll
  for (int d = 0; d < DH; ++d) m = fmaxf(m, qv[d]);
  float s = 0.f;
#pragma unroll
  for (int d = 0; d < DH; ++d) s += __expf(qv[d] - m);
  const float inv = QSCALE / s;
  v16h a, b0, b1;
#pragma unroll
  for (int j = 0; j < 16; ++j) {
    const int d = kmap16(j, khalf);
    a[j]  = (_Float16)(__expf(qv[d] - m) * inv);
    const size_t coff = (size_t)bh * (DH * DH) + d * DH + mr;
    b0[j] = (_Float16)ctx[coff];
    b1[j] = (_Float16)ctx[coff + 16];
  }
  v8f acc0 = {}, acc1 = {};
  acc0 = __builtin_amdgcn_wmma_f32_16x16x32_f16(false, a, false, b0, (short)0, acc0, false, false);
  acc1 = __builtin_amdgcn_wmma_f32_16x16x32_f16(false, a, false, b1, (short)0, acc1, false, false);
#pragma unroll
  for (int r = 0; r < 8; ++r) {
    const int tk = tt * 16 + r + 8 * khalf;
    const size_t base = (size_t)(b * NTOK + tk) * HID + h * DH + mr;
    O[base]      = acc0[r];
    O[base + 16] = acc1[r];
  }
}

// ---------------- K5: Y = LN(O @ Wout + b) * gamma + beta ----------------
// block = 16 rows x 256 cols; 8 waves, each owning 2 col-tiles with a shared
// A fragment per k-step; LN via wave32 shuffles.
__global__ void la_out_ln(const float* __restrict__ O,
                          const float* __restrict__ wout,
                          const float* __restrict__ bout,
                          const float* __restrict__ gamma,
                          const float* __restrict__ beta,
                          float* __restrict__ out) {
  __shared__ float tile[16][C_ + 4];
  __shared__ float mu_s[16], rs_s[16];
  const int rowbase = blockIdx.x * 16;
  const int tid  = threadIdx.x;
  const int wave = tid >> 5;
  const int lane = tid & 31;
  const int mr = lane & 15, khalf = lane >> 4;
  const int colbase = wave * 32;
  const float* arow = O + (size_t)(rowbase + mr) * HID;

  v8f acc0 = {}, acc1 = {};
  for (int k0 = 0; k0 < HID; k0 += 32) {
    const v16h a = load_a_frag(arow, k0, khalf);
    v16h b0, b1;
#pragma unroll
    for (int j = 0; j < 16; ++j) {
      const size_t woff = (size_t)(k0 + kmap16(j, khalf)) * C_ + colbase + mr;
      b0[j] = (_Float16)wout[woff];
      b1[j] = (_Float16)wout[woff + 16];
    }
    acc0 = __builtin_amdgcn_wmma_f32_16x16x32_f16(false, a, false, b0, (short)0, acc0, false, false);
    acc1 = __builtin_amdgcn_wmma_f32_16x16x32_f16(false, a, false, b1, (short)0, acc1, false, false);
  }
#pragma unroll
  for (int r = 0; r < 8; ++r) {
    const int row = r + 8 * khalf;
    const int col = colbase + mr;
    tile[row][col]      = acc0[r] + bout[col];
    tile[row][col + 16] = acc1[r] + bout[col + 16];
  }
  __syncthreads();

  // LayerNorm stats: wave w reduces rows 2w, 2w+1 via lane shuffles (wave32)
  for (int rr = 0; rr < 2; ++rr) {
    const int row = wave * 2 + rr;
    float s = 0.f, s2 = 0.f;
    for (int cidx = lane; cidx < C_; cidx += 32) {
      const float v = tile[row][cidx];
      s += v; s2 += v * v;
    }
#pragma unroll
    for (int off = 16; off > 0; off >>= 1) {
      s  += __shfl_xor(s,  off, 32);
      s2 += __shfl_xor(s2, off, 32);
    }
    if (lane == 0) {
      const float mu  = s  * (1.f / C_);
      const float var = s2 * (1.f / C_) - mu * mu;
      mu_s[row] = mu;
      rs_s[row] = rsqrtf(var + LN_EPS);
    }
  }
  __syncthreads();

  const int col = tid;                       // 0..255
  const float g = gamma[col], be = beta[col];
#pragma unroll
  for (int r = 0; r < 16; ++r) {
    const float v = tile[r][col];
    out[(size_t)(rowbase + r) * C_ + col] = (v - mu_s[r]) * rs_s[r] * g + be;
  }
}

extern "C" void kernel_launch(void* const* d_in, const int* in_sizes, int n_in,
                              void* d_out, int out_size, void* d_ws, size_t ws_size,
                              hipStream_t stream) {
  const float* x     = (const float*)d_in[0];  // [16,64,64,256]
  const float* w_qkv = (const float*)d_in[1];  // [256,384]
  const float* w_out = (const float*)d_in[2];  // [128,256]
  const float* b_out = (const float*)d_in[3];  // [256]
  const float* gamma = (const float*)d_in[4];  // [256]
  const float* beta  = (const float*)d_in[5];  // [256]
  float* out = (float*)d_out;

  float* ws    = (float*)d_ws;
  float* qkv   = ws;                                   // 65536*384
  float* O     = qkv  + (size_t)NROWS * F3;            // 65536*128
  float* kmaxb = O    + (size_t)NROWS * HID;           // 2048
  float* ksumb = kmaxb + (size_t)HEADS * BATCH * DH;   // 2048
  float* ctx   = ksumb + (size_t)HEADS * BATCH * DH;   // 64*32*32

  la_qkv_gemm    <<<dim3(NROWS / 16, F3 / 64), 32, 0, stream>>>(x, w_qkv, qkv);
  la_ksm_stats   <<<dim3(BATCH * HEADS), 256, 0, stream>>>(qkv, kmaxb, ksumb);
  la_context_gemm<<<dim3(BATCH * HEADS, 2), 32, 0, stream>>>(qkv, kmaxb, ksumb, ctx);
  la_attn_out    <<<dim3(NTOK / 16, BATCH * HEADS), 32, 0, stream>>>(qkv, ctx, O);
  la_out_ln      <<<dim3(NROWS / 16), 256, 0, stream>>>(O, w_out, b_out, gamma, beta, out);
}
